// SetConvLayer_59098749993249
// MI455X (gfx1250) — matrix operand order repaired and checked by reading, not compiled, not run on hardware
//
#include <hip/hip_runtime.h>
#include <hip/hip_bf16.h>

// Problem constants (match reference).
#define BB 4
#define NN 8192
#define FF 64
#define MM 2048
#define KK 32
#define HH 64
#define RR 0.1f
#define KP 96          // F+3 = 67 padded up to 3 x 32 for WMMA K-steps
#define CAND 64        // candidate cap per center (expected ~34 within radius)
#define CG  8          // centers per MLP block (amortizes LDS weight staging)

typedef __attribute__((ext_vector_type(16))) _Float16 v16h;
typedef __attribute__((ext_vector_type(8)))  float    v8f;

// ---------------------------------------------------------------------------
// Kernel 0: convert W1 [67,64] / W2 [64,64] (f32, K-major) to f16, N-major
// (K-contiguous) with zero padding of the K dim of W1 to 96.
// ---------------------------------------------------------------------------
__global__ void convert_weights_kernel(const float* __restrict__ W1,
                                       const float* __restrict__ W2,
                                       _Float16* __restrict__ W1h,
                                       _Float16* __restrict__ W2h) {
  int t = blockIdx.x * blockDim.x + threadIdx.x;
  if (t < HH * KP) {                   // W1h[n*96 + k]
    int n = t / KP, k = t % KP;
    W1h[t] = (k < FF + 3) ? (_Float16)W1[k * HH + n] : (_Float16)0.0f;
  }
  if (t < HH * HH) {                   // W2h[n*64 + k]
    int n = t / HH, k = t % HH;
    W2h[t] = (_Float16)W2[k * HH + n];
  }
}

// ---------------------------------------------------------------------------
// Kernel 1: farthest point sampling. One 1024-thread block per cloud.
// Each thread keeps 8 points' min-distance in registers; block argmax via
// wave32 shuffles + LDS. Writes centers and new_batch directly into d_out.
// ---------------------------------------------------------------------------
__global__ void __launch_bounds__(1024)
fps_kernel(const float* __restrict__ pos,
           float* __restrict__ centers,     // [B*M,3] region of d_out
           int* __restrict__ new_batch) {   // [B*M]   region of d_out
  const int b = blockIdx.x;
  const int t = threadIdx.x;
  const int lane = t & 31;
  const int wv = t >> 5;                    // 32 waves
  const float* pc = pos + (size_t)b * NN * 3;

  __shared__ float sLast[3];
  __shared__ float sRedV[32];
  __shared__ int   sRedI[32];
  __shared__ int   sCur;

  float px[8], py[8], pz[8], md[8];
#pragma unroll
  for (int i = 0; i < 8; ++i) {
    int p = t + i * 1024;
    px[i] = pc[p * 3 + 0];
    py[i] = pc[p * 3 + 1];
    pz[i] = pc[p * 3 + 2];
    md[i] = 1e30f;
  }

  int cur = 0;
  for (int m = 0; m < MM; ++m) {
    if (t == 0) {
      float cx = pc[cur * 3 + 0], cy = pc[cur * 3 + 1], cz = pc[cur * 3 + 2];
      size_t row = (size_t)b * MM + m;
      centers[row * 3 + 0] = cx;
      centers[row * 3 + 1] = cy;
      centers[row * 3 + 2] = cz;
      sLast[0] = cx; sLast[1] = cy; sLast[2] = cz;
    }
    __syncthreads();
    float lx = sLast[0], ly = sLast[1], lz = sLast[2];

    float bv = -1.0f; int bi = 0x7fffffff;
#pragma unroll
    for (int i = 0; i < 8; ++i) {
      float dx = px[i] - lx, dy = py[i] - ly, dz = pz[i] - lz;
      float d = dx * dx + dy * dy + dz * dz;
      md[i] = fminf(md[i], d);
      int p = t + i * 1024;
      if (md[i] > bv || (md[i] == bv && p < bi)) { bv = md[i]; bi = p; }
    }
    // wave32 argmax (lowest-index tiebreak, like jnp.argmax)
#pragma unroll
    for (int off = 16; off > 0; off >>= 1) {
      float ov = __shfl_xor(bv, off, 32);
      int   oi = __shfl_xor(bi, off, 32);
      if (ov > bv || (ov == bv && oi < bi)) { bv = ov; bi = oi; }
    }
    if (lane == 0) { sRedV[wv] = bv; sRedI[wv] = bi; }
    __syncthreads();
    if (wv == 0) {
      float v = sRedV[lane]; int i = sRedI[lane];
#pragma unroll
      for (int off = 16; off > 0; off >>= 1) {
        float ov = __shfl_xor(v, off, 32);
        int   oi = __shfl_xor(i, off, 32);
        if (ov > v || (ov == v && oi < i)) { v = ov; i = oi; }
      }
      if (lane == 0) sCur = i;
    }
    __syncthreads();
    cur = sCur;
  }

  // new_batch ids
  for (int j = t; j < MM; j += 1024) new_batch[b * MM + j] = b;
}

// ---------------------------------------------------------------------------
// Kernel 2: radius neighbor search, one wave per center. Append candidates
// within R via LDS atomics, then wave-parallel selection of the K nearest
// (ascending distance, index tiebreak). Invalid slots get index 0.
// ---------------------------------------------------------------------------
__global__ void __launch_bounds__(256)
radius_kernel(const float* __restrict__ pos,
              const float* __restrict__ centers,
              int* __restrict__ nbr,        // [B*M,K]
              int* __restrict__ cnt) {      // [B*M]
  const int t = threadIdx.x;
  const int lane = t & 31;
  const int w = t >> 5;                     // 8 waves / block
  const int g = blockIdx.x * 8 + w;         // center id
  const int b = g / MM;
  const float* pc = pos + (size_t)b * NN * 3;

  __shared__ float sD[8][CAND];
  __shared__ int   sI[8][CAND];
  __shared__ int   sCnt[8];

  if (lane == 0) sCnt[w] = 0;
  __syncthreads();

  const float cx = centers[g * 3 + 0];
  const float cy = centers[g * 3 + 1];
  const float cz = centers[g * 3 + 2];
  const float r2 = RR * RR;

  for (int p = lane; p < NN; p += 32) {
    float dx = pc[p * 3 + 0] - cx;
    float dy = pc[p * 3 + 1] - cy;
    float dz = pc[p * 3 + 2] - cz;
    float d2 = dx * dx + dy * dy + dz * dz;
    if (d2 <= r2) {
      int slot = atomicAdd(&sCnt[w], 1);
      if (slot < CAND) { sD[w][slot] = d2; sI[w][slot] = p; }
    }
  }
  __syncthreads();

  int n = sCnt[w]; n = n < CAND ? n : CAND;
  int valid = 0;
  for (int k = 0; k < KK; ++k) {
    float bv = 1e30f; int bj = 0x7fffffff;
    for (int j = lane; j < n; j += 32) {
      float d = sD[w][j];
      if (d < bv || (d == bv && j < bj)) { bv = d; bj = j; }
    }
#pragma unroll
    for (int off = 16; off > 0; off >>= 1) {
      float ov = __shfl_xor(bv, off, 32);
      int   oj = __shfl_xor(bj, off, 32);
      if (ov < bv || (ov == bv && oj < bj)) { bv = ov; bj = oj; }
    }
    if (bv < 1e29f) {
      ++valid;
      if (lane == 0) { nbr[(size_t)g * KK + k] = sI[w][bj]; sD[w][bj] = 1e30f; }
    } else if (lane == 0) {
      nbr[(size_t)g * KK + k] = 0;
    }
  }
  if (lane == 0) cnt[g] = valid;
}

// ---------------------------------------------------------------------------
// WMMA fragment loaders (CDNA5 wave32 16-bit layouts, CDNA5 ISA).
// A (16x32, row-major LDS): lane%16 -> row; lane/16 selects the two 8-half
// K runs {sel*8..+7} and {16+sel*8..+7}.
// B (32x16 from N-major LDS): lane%16 -> col; lane/16 selects the same two
// 8-half K runs as A.
// ---------------------------------------------------------------------------
__device__ __forceinline__ v16h load_frag_a(const _Float16* m, int rowStride,
                                            int r0, int k0, int lane) {
  int row = r0 + (lane & 15);
  int sel = lane >> 4;
  const _Float16* p = m + (size_t)row * rowStride + k0 + sel * 8;
  v16h a;
#pragma unroll
  for (int i = 0; i < 8; ++i) a[i] = p[i];
#pragma unroll
  for (int i = 0; i < 8; ++i) a[8 + i] = p[16 + i];
  return a;
}

__device__ __forceinline__ v16h load_frag_b(const _Float16* wt, int kStride,
                                            int n0, int k0, int lane) {
  int col = n0 + (lane & 15);
  int sel = lane >> 4;
  const _Float16* p = wt + (size_t)col * kStride + k0 + sel * 8;
  v16h bfrag;
#pragma unroll
  for (int i = 0; i < 8; ++i) bfrag[i] = p[i];
#pragma unroll
  for (int i = 0; i < 8; ++i) bfrag[8 + i] = p[16 + i];
  return bfrag;
}

// ---------------------------------------------------------------------------
// Kernel 3: PointConv MLP + masked max-aggregation. One 8-wave block handles
// CG=8 centers (weights staged into LDS once, reused 8x). Per center, wave w
// owns output tile (mtile = w/4, ntile = w%4) of the [32,64] activation.
// Layer1: [32x96]f16 @ [96x64]f16 (3 WMMA K-steps), ReLU;
// Layer2: [32x64] @ [64x64] (2 K-steps), ReLU; masked row-max -> out[g, :].
// ---------------------------------------------------------------------------
__global__ void __launch_bounds__(256)
mlp_kernel(const float* __restrict__ x,
           const float* __restrict__ pos,
           const float* __restrict__ centers,
           const int* __restrict__ nbr,
           const int* __restrict__ cnt,
           const float* __restrict__ b1,
           const float* __restrict__ b2,
           const _Float16* __restrict__ W1h,
           const _Float16* __restrict__ W2h,
           float* __restrict__ out) {
  __shared__ __align__(16) _Float16 sW1[HH * KP];     // 12 KB, N-major
  __shared__ __align__(16) _Float16 sW2[HH * HH];     //  8 KB, N-major
  __shared__ __align__(16) _Float16 sFeat[KK * KP];   //  6 KB, row-major
  __shared__ __align__(16) _Float16 sH1[KK * HH];     //  4 KB, row-major
  __shared__ float sRed[HH * 4];                      //  1 KB
  __shared__ int   sNbr[KK];

  const int t = threadIdx.x;
  const int lane = t & 31;
  const int w = t >> 5;
  const int mtile = w >> 2;                 // 0..1 (neighbor rows)
  const int ntile = w & 3;                  // 0..3 (feature cols)
  const int col = ntile * 16 + (lane & 15);
  const int sel = lane >> 4;

  // Stage f16 weights (L2-resident) into LDS once per block.
  {
    const uint4* s1 = (const uint4*)W1h;
    uint4* d1 = (uint4*)sW1;
    for (int i = t; i < (HH * KP) / 8; i += 256) d1[i] = s1[i];
    const uint4* s2 = (const uint4*)W2h;
    uint4* d2 = (uint4*)sW2;
    for (int i = t; i < (HH * HH) / 8; i += 256) d2[i] = s2[i];
  }

  const float bias1 = b1[col];
  const float bias2 = b2[col];

  for (int c = 0; c < CG; ++c) {
    const int g = blockIdx.x * CG + c;      // center id in [0, B*M)
    const int b = g / MM;
    const float* pcb = pos + (size_t)b * NN * 3;
    const float* xb  = x   + (size_t)b * NN * FF;

    // Neighbor list for this center -> LDS (read once, reused ~12x each).
    if (t < KK) sNbr[t] = nbr[(size_t)g * KK + t];
    __syncthreads();

    // Gather features: [x_j (64) | pos_j - pos_i (3) | zeros (29)] -> f16.
    const float cx = centers[g * 3 + 0];
    const float cy = centers[g * 3 + 1];
    const float cz = centers[g * 3 + 2];
    for (int i = t; i < KK * KP; i += 256) {
      int row = i / KP, cc = i % KP;
      int nb = sNbr[row];
      float v;
      if (cc < FF) {
        v = xb[(size_t)nb * FF + cc];
      } else if (cc < FF + 3) {
        int d = cc - FF;
        float ctr = (d == 0) ? cx : ((d == 1) ? cy : cz);
        v = pcb[(size_t)nb * 3 + d] - ctr;
      } else {
        v = 0.0f;
      }
      sFeat[i] = (_Float16)v;
    }
    __syncthreads();

    // ---- Layer 1: feat[32,96] @ W1[96,64] ----
    v8f acc = {};
#pragma unroll
    for (int kc = 0; kc < 3; ++kc) {
      v16h a  = load_frag_a(sFeat, KP, mtile * 16, kc * 32, lane);
      v16h bf = load_frag_b(sW1,  KP, ntile * 16, kc * 32, lane);
      acc = __builtin_amdgcn_wmma_f32_16x16x32_f16(
          false, a, false, bf, (short)0, acc, false, false);
    }
#pragma unroll
    for (int r = 0; r < 8; ++r) {
      float v = acc[r] + bias1;
      v = v > 0.0f ? v : 0.0f;
      int row = mtile * 16 + sel * 8 + r;   // C/D layout: VGPR r -> M=r+8*sel
      sH1[row * HH + col] = (_Float16)v;
    }
    __syncthreads();

    // ---- Layer 2: h1[32,64] @ W2[64,64] ----
    v8f acc2 = {};
#pragma unroll
    for (int kc = 0; kc < 2; ++kc) {
      v16h a  = load_frag_a(sH1, HH, mtile * 16, kc * 32, lane);
      v16h bf = load_frag_b(sW2, HH, ntile * 16, kc * 32, lane);
      acc2 = __builtin_amdgcn_wmma_f32_16x16x32_f16(
          false, a, false, bf, (short)0, acc2, false, false);
    }

    // ---- Bias + ReLU + validity-masked max over neighbor rows ----
    {
      int nvalid = cnt[g];
      float mx = -1e30f;
#pragma unroll
      for (int r = 0; r < 8; ++r) {
        int row = mtile * 16 + sel * 8 + r;
        float v = acc2[r] + bias2;
        v = v > 0.0f ? v : 0.0f;
        if (row < nvalid) mx = fmaxf(mx, v);
      }
      sRed[col * 4 + mtile * 2 + sel] = mx; // each of 256 slots written once
    }
    __syncthreads();

    if (t < HH) {
      float m0 = sRed[t * 4 + 0];
      m0 = fmaxf(m0, sRed[t * 4 + 1]);
      m0 = fmaxf(m0, sRed[t * 4 + 2]);
      m0 = fmaxf(m0, sRed[t * 4 + 3]);
      if (!(m0 > -1e29f)) m0 = 0.0f;        // no valid neighbor -> 0
      out[(size_t)g * HH + t] = m0;
    }
    __syncthreads();                        // before reusing sFeat/sNbr
  }
}

// ---------------------------------------------------------------------------
// Launch. Inputs (setup_inputs order): x, pos, batch, W1, b1, W2, b2.
// d_out (float): [out B*M*H | centers B*M*3 | new_batch B*M (int bits)].
// ---------------------------------------------------------------------------
extern "C" void kernel_launch(void* const* d_in, const int* in_sizes, int n_in,
                              void* d_out, int out_size, void* d_ws, size_t ws_size,
                              hipStream_t stream) {
  (void)in_sizes; (void)n_in; (void)out_size; (void)ws_size;
  const float* x   = (const float*)d_in[0];
  const float* pos = (const float*)d_in[1];
  // d_in[2] = batch (unused: clouds are contiguous, equal-sized)
  const float* W1  = (const float*)d_in[3];
  const float* b1  = (const float*)d_in[4];
  const float* W2  = (const float*)d_in[5];
  const float* b2  = (const float*)d_in[6];

  float* out_feat   = (float*)d_out;                       // [B*M, H]
  float* out_ctr    = out_feat + (size_t)BB * MM * HH;     // [B*M, 3]
  int*   out_batch  = (int*)(out_ctr + (size_t)BB * MM * 3);

  char* ws = (char*)d_ws;
  int*      d_nbr = (int*)ws;                               // B*M*K ints (1 MB)
  int*      d_cnt = (int*)(ws + (size_t)BB * MM * KK * 4);  // B*M ints
  _Float16* d_W1h = (_Float16*)(ws + (size_t)BB * MM * KK * 4 + (size_t)BB * MM * 4);
  _Float16* d_W2h = d_W1h + HH * KP;

  convert_weights_kernel<<<(HH * KP + 255) / 256, 256, 0, stream>>>(W1, W2, d_W1h, d_W2h);
  fps_kernel<<<BB, 1024, 0, stream>>>(pos, out_ctr, out_batch);
  radius_kernel<<<(BB * MM) / 8, 256, 0, stream>>>(pos, out_ctr, d_nbr, d_cnt);
  mlp_kernel<<<(BB * MM) / CG, 256, 0, stream>>>(x, pos, out_ctr, d_nbr, d_cnt,
                                                 b1, b2, d_W1h, d_W2h, out_feat);
}
